// SSMBlock_1975684956292
// MI455X (gfx1250) — matrix operand order repaired and checked by reading, not compiled, not run on hardware
//
#include <hip/hip_runtime.h>
#include <hip/hip_bf16.h>

typedef __attribute__((ext_vector_type(16))) _Float16 v16h;
typedef __attribute__((ext_vector_type(8)))  _Float16 v8h;
typedef __attribute__((ext_vector_type(8)))  float    v8f;
typedef __attribute__((ext_vector_type(4)))  int      i32x4;

#define D_MODEL 1024
#define D_STATE 16
#define D_CONVK 4
#define D_INNER 2048
#define BATCH 2
#define SEQ 4096
#define ROWS (BATCH*SEQ)   /* 8192 */
#define NCHUNK 64
#define CLEN 64
#define LOG2E 1.44269504f
#define KSTEP 64

// Async global->LDS copy path (CDNA5 GLOBAL_LOAD_ASYNC_TO_LDS_*, ASYNCcnt).
#if defined(__has_builtin)
#if __has_builtin(__builtin_amdgcn_global_load_async_to_lds_b128) && \
    __has_builtin(__builtin_amdgcn_s_wait_asynccnt)
#define USE_ASYNC 1
#endif
#endif
#ifndef USE_ASYNC
#define USE_ASYNC 0
#endif

#if USE_ASYNC
typedef __attribute__((address_space(1))) i32x4 gi32x4;   // global 128b chunk
typedef __attribute__((address_space(3))) i32x4 li32x4;   // LDS 128b chunk
#endif

// ---------------------------------------------------------------------------
// Weight convert + transpose: src f32 (K x N, row-major) -> dst f16 (N x K).
// ---------------------------------------------------------------------------
__global__ void convT(const float* __restrict__ src, _Float16* __restrict__ dst,
                      int K, int N) {
  int idx = blockIdx.x * 256 + threadIdx.x;   // grid sized exactly K*N/256
  int k = idx / N, n = idx % N;
  dst[(size_t)n * K + k] = (_Float16)src[idx];
}

// ---------------------------------------------------------------------------
// LayerNorm over D_MODEL, emit f16 activations (WMMA A operand).
// ---------------------------------------------------------------------------
__global__ void ln_to_f16(const float* __restrict__ x, const float* __restrict__ gamma,
                          const float* __restrict__ beta, _Float16* __restrict__ xn16) {
  __shared__ float s1[256], s2[256];
  int row = blockIdx.x, t = threadIdx.x;
  const float* xr = x + (size_t)row * D_MODEL;
  float v[4], sum = 0.f, sq = 0.f;
#pragma unroll
  for (int j = 0; j < 4; j++) { v[j] = xr[t + 256 * j]; sum += v[j]; sq += v[j] * v[j]; }
  s1[t] = sum; s2[t] = sq; __syncthreads();
  for (int s = 128; s > 0; s >>= 1) {
    if (t < s) { s1[t] += s1[t + s]; s2[t] += s2[t + s]; }
    __syncthreads();
  }
  float mu = s1[0] * (1.f / D_MODEL);
  float var = s2[0] * (1.f / D_MODEL) - mu * mu;
  float rs = rsqrtf(var + 1e-5f);
#pragma unroll
  for (int j = 0; j < 4; j++) {
    int i = t + 256 * j;
    xn16[(size_t)row * D_MODEL + i] = (_Float16)((v[j] - mu) * rs * gamma[i] + beta[i]);
  }
}

// ---------------------------------------------------------------------------
// WMMA GEMM: C(M x N, f32) = A(M x K, f16 row-major) * Bt(N x K, f16)^T [+resid]
// Block 256 thr = 8 waves; block tile 128x128; wave tile 64x32 (4x2 frags).
// KSTEP=64, double-buffered LDS; async global->LDS copies when available.
// ---------------------------------------------------------------------------
__global__ void __launch_bounds__(256)
wmma_gemm(const _Float16* __restrict__ Ag, const _Float16* __restrict__ Btg,
          float* __restrict__ Cg, const float* __restrict__ resid,
          int M, int N, int K) {
  __shared__ __align__(32) _Float16 As[2][128 * KSTEP];
  __shared__ __align__(32) _Float16 Bs[2][128 * KSTEP];
  const int t = threadIdx.x;
  const int lane = t & 31, w = t >> 5;
  const int wm = w >> 2, wn = w & 3;               // 2x4 wave grid
  const int mBase = blockIdx.x * 128, nBase = blockIdx.y * 128;
  const int ln = lane & 15, half = lane >> 4;

  v8f acc[4][2] = {};

  // Tile copy: 128 rows x 64 halfs (128B/row) = 1024 chunks of 16B; 4/thread
  // per matrix (8 async ops outstanding per thread per tile).
  auto issueTile = [&](int buf, int k0) {
#pragma unroll
    for (int i = 0; i < 4; i++) {
      int c = t + 256 * i;
      int r = c >> 3, off = c & 7;
#if USE_ASYNC
      __builtin_amdgcn_global_load_async_to_lds_b128(
          (gi32x4*)(Ag + (size_t)(mBase + r) * K + k0 + off * 8),
          (li32x4*)&As[buf][r * KSTEP + off * 8], 0, 0);
      __builtin_amdgcn_global_load_async_to_lds_b128(
          (gi32x4*)(Btg + (size_t)(nBase + r) * K + k0 + off * 8),
          (li32x4*)&Bs[buf][r * KSTEP + off * 8], 0, 0);
#else
      *(v8h*)&As[buf][r * KSTEP + off * 8] =
          *(const v8h*)(Ag + (size_t)(mBase + r) * K + k0 + off * 8);
      *(v8h*)&Bs[buf][r * KSTEP + off * 8] =
          *(const v8h*)(Btg + (size_t)(nBase + r) * K + k0 + off * 8);
#endif
    }
  };

  const int nIter = K / KSTEP;
  issueTile(0, 0);
  for (int i = 0; i < nIter; i++) {
    if (i + 1 < nIter) {
      issueTile((i + 1) & 1, (i + 1) * KSTEP);     // prefetch next tile
#if USE_ASYNC
      __builtin_amdgcn_s_wait_asynccnt(8);         // tile i complete; i+1 inflight
#endif
    } else {
#if USE_ASYNC
      __builtin_amdgcn_s_wait_asynccnt(0);
#endif
    }
    __syncthreads();

    const _Float16* Ab = As[i & 1];
    const _Float16* Bb = Bs[i & 1];
#pragma unroll
    for (int k32 = 0; k32 < 2; k32++) {
      v16h af[4], bf[2];
#pragma unroll
      for (int mi = 0; mi < 4; mi++) {             // A: k = (i&7)+8*half+16*(i>>3)
        int r = wm * 64 + mi * 16 + ln;
        v8h lo = *(const v8h*)&Ab[r * KSTEP + k32 * 32 + 8 * half];
        v8h hi = *(const v8h*)&Ab[r * KSTEP + k32 * 32 + 16 + 8 * half];
        af[mi] = __builtin_shufflevector(lo, hi, 0, 1, 2, 3, 4, 5, 6, 7,
                                         8, 9, 10, 11, 12, 13, 14, 15);
      }
#pragma unroll
      for (int ni = 0; ni < 2; ni++) {             // B: k = (i&7)+8*half+16*(i>>3)
        int r = wn * 32 + ni * 16 + ln;
        v8h lo = *(const v8h*)&Bb[r * KSTEP + k32 * 32 + 8 * half];
        v8h hi = *(const v8h*)&Bb[r * KSTEP + k32 * 32 + 16 + 8 * half];
        bf[ni] = __builtin_shufflevector(lo, hi, 0, 1, 2, 3, 4, 5, 6, 7,
                                         8, 9, 10, 11, 12, 13, 14, 15);
      }
#pragma unroll
      for (int mi = 0; mi < 4; mi++)
#pragma unroll
        for (int ni = 0; ni < 2; ni++)
          acc[mi][ni] = __builtin_amdgcn_wmma_f32_16x16x32_f16(
              false, af[mi], false, bf[ni], (short)0, acc[mi][ni], false, false);
    }
    __syncthreads();   // all waves done with buf (i&1) before it is re-filled
  }

#pragma unroll
  for (int mi = 0; mi < 4; mi++)
    for (int ni = 0; ni < 2; ni++)
#pragma unroll
      for (int r = 0; r < 8; r++) {                // C/D: M=r+8*half, N=ln
        int row = mBase + wm * 64 + mi * 16 + r + 8 * half;
        int col = nBase + wn * 32 + ni * 16 + ln;
        float v = acc[mi][ni][r];
        if (resid) v += resid[(size_t)row * N + col];
        Cg[(size_t)row * N + col] = v;
      }
}

// ---------------------------------------------------------------------------
// Depthwise causal conv (k=4) + bias + SiLU. x_inner = xz[:, 0:2048].
// ---------------------------------------------------------------------------
__global__ void conv_silu(const float* __restrict__ xz, const float* __restrict__ cw,
                          const float* __restrict__ cb, float* __restrict__ xconv) {
  int idx = blockIdx.x * 256 + threadIdx.x;
  int d = idx & (D_INNER - 1);
  int row = idx >> 11;
  int l = row & (SEQ - 1);
  int b = row >> 12;
  float acc = cb[d];
#pragma unroll
  for (int j = 0; j < D_CONVK; j++) {
    int ll = l - (D_CONVK - 1) + j;
    if (ll >= 0)
      acc += cw[d * D_CONVK + j] * xz[(size_t)(b * SEQ + ll) * (2 * D_INNER) + d];
  }
  xconv[(size_t)row * D_INNER + d] = acc * (1.f / (1.f + __expf(-acc)));
}

// ---------------------------------------------------------------------------
// xp = x_conv @ W_x (K=2048, N=33) -> B[16], C[16], delta=softplus(col 32).
// ---------------------------------------------------------------------------
__global__ void proj_x(const float* __restrict__ xconv, const float* __restrict__ Wx,
                       float* __restrict__ Bm, float* __restrict__ Cm,
                       float* __restrict__ dl) {
  __shared__ float xr[D_INNER];
  __shared__ float part[33 * 4];
  int row = blockIdx.x, t = threadIdx.x;
#pragma unroll
  for (int j = 0; j < D_INNER / 256; j++)
    xr[t + 256 * j] = xconv[(size_t)row * D_INNER + t + 256 * j];
  __syncthreads();
  if (t < 132) {
    int o = t >> 2, q = t & 3;
    float s = 0.f;
    for (int k = q * 512; k < q * 512 + 512; k++) s += xr[k] * Wx[k * 33 + o];
    part[t] = s;
  }
  __syncthreads();
  if (t < 33) {
    float v = part[4 * t] + part[4 * t + 1] + part[4 * t + 2] + part[4 * t + 3];
    if (t < 16)      Bm[row * 16 + t] = v;
    else if (t < 32) Cm[row * 16 + (t - 16)] = v;
    else             dl[row] = (v > 20.f) ? v : logf(1.f + __expf(v));
  }
}

// ---------------------------------------------------------------------------
// Chunked selective scan, pass 1: per-chunk summaries (h0=0 local state S,
// chunk decay product P). Whole chunk's B/dt preloaded to LDS, one barrier.
// ---------------------------------------------------------------------------
__global__ void scan_chunk_summary(const float* __restrict__ xconv,
                                   const float* __restrict__ Bm,
                                   const float* __restrict__ dl,
                                   const float* __restrict__ A_log,
                                   float* __restrict__ S, float* __restrict__ P) {
  int blk = blockIdx.x;                 // b*512 + c*8 + dblk
  int dblk = blk & 7;
  int c = (blk >> 3) & (NCHUNK - 1);
  int b = blk >> 9;
  int t = threadIdx.x;
  int d = dblk * 256 + t;
  __shared__ float sB[CLEN * 16];
  __shared__ float sdt[CLEN];
  int l0 = c * CLEN;
  for (int j = t; j < CLEN * 16; j += 256) sB[j] = Bm[(size_t)(b * SEQ + l0) * 16 + j];
  if (t < CLEN) sdt[t] = dl[b * SEQ + l0 + t];
  __syncthreads();

  float An[16], h[16], Pr[16];
#pragma unroll
  for (int n = 0; n < 16; n++) {
    An[n] = -__expf(A_log[d * 16 + n]) * LOG2E;   // exp(dt*A) = exp2(dt*An)
    h[n] = 0.f; Pr[n] = 1.f;
  }
  for (int li = 0; li < CLEN; li++) {
    int row = b * SEQ + l0 + li;
    float dt = sdt[li];
    float dx = dt * xconv[(size_t)row * D_INNER + d];
    const float* Bl = &sB[li * 16];
#pragma unroll
    for (int n = 0; n < 16; n++) {
      float e = exp2f(dt * An[n]);
      Pr[n] *= e;
      h[n] = e * h[n] + dx * Bl[n];
    }
  }
  size_t base = ((size_t)(b * NCHUNK + c) * D_INNER + d) * 16;
#pragma unroll
  for (int n = 0; n < 16; n++) { S[base + n] = h[n]; P[base + n] = Pr[n]; }
}

// Pass 2: sequential prefix over 64 chunks (parallel over b*d = 4096).
__global__ void scan_chunk_prefix(const float* __restrict__ S,
                                  const float* __restrict__ P,
                                  float* __restrict__ Hinit) {
  int idx = blockIdx.x * 256 + threadIdx.x;
  int b = idx >> 11, d = idx & (D_INNER - 1);
  float h[16];
#pragma unroll
  for (int n = 0; n < 16; n++) h[n] = 0.f;
  for (int c = 0; c < NCHUNK; c++) {
    size_t base = ((size_t)(b * NCHUNK + c) * D_INNER + d) * 16;
#pragma unroll
    for (int n = 0; n < 16; n++) Hinit[base + n] = h[n];
#pragma unroll
    for (int n = 0; n < 16; n++) h[n] = P[base + n] * h[n] + S[base + n];
  }
}

// Pass 3: final scan from corrected initial states; fuse D-skip, SiLU(z)
// gating, and f16 conversion (A operand of the output GEMM).
__global__ void scan_final(const float* __restrict__ xconv, const float* __restrict__ xz,
                           const float* __restrict__ Bm, const float* __restrict__ Cm,
                           const float* __restrict__ dl, const float* __restrict__ A_log,
                           const float* __restrict__ Dskip, const float* __restrict__ Hinit,
                           _Float16* __restrict__ yg16) {
  int blk = blockIdx.x;
  int dblk = blk & 7, c = (blk >> 3) & (NCHUNK - 1), b = blk >> 9;
  int t = threadIdx.x, d = dblk * 256 + t;
  __shared__ float sB[CLEN * 16], sC[CLEN * 16];
  __shared__ float sdt[CLEN];
  int l0 = c * CLEN;
  for (int j = t; j < CLEN * 16; j += 256) {
    sB[j] = Bm[(size_t)(b * SEQ + l0) * 16 + j];
    sC[j] = Cm[(size_t)(b * SEQ + l0) * 16 + j];
  }
  if (t < CLEN) sdt[t] = dl[b * SEQ + l0 + t];
  __syncthreads();

  float An[16], h[16];
#pragma unroll
  for (int n = 0; n < 16; n++) An[n] = -__expf(A_log[d * 16 + n]) * LOG2E;
  size_t base = ((size_t)(b * NCHUNK + c) * D_INNER + d) * 16;
#pragma unroll
  for (int n = 0; n < 16; n++) h[n] = Hinit[base + n];
  float dsk = Dskip[d];
  for (int li = 0; li < CLEN; li++) {
    int row = b * SEQ + l0 + li;
    float dt = sdt[li];
    float xv = xconv[(size_t)row * D_INNER + d];
    float dx = dt * xv;
    const float* Bl = &sB[li * 16];
    const float* Cl = &sC[li * 16];
    float y = 0.f;
#pragma unroll
    for (int n = 0; n < 16; n++) {
      float e = exp2f(dt * An[n]);
      h[n] = e * h[n] + dx * Bl[n];
      y += h[n] * Cl[n];
    }
    y += xv * dsk;
    float z = xz[(size_t)row * (2 * D_INNER) + D_INNER + d];
    y *= z * (1.f / (1.f + __expf(-z)));
    yg16[(size_t)row * D_INNER + d] = (_Float16)y;
  }
}

// ---------------------------------------------------------------------------
extern "C" void kernel_launch(void* const* d_in, const int* in_sizes, int n_in,
                              void* d_out, int out_size, void* d_ws, size_t ws_size,
                              hipStream_t stream) {
  (void)in_sizes; (void)n_in; (void)out_size; (void)ws_size;
  const float* x     = (const float*)d_in[0];
  const float* W_in  = (const float*)d_in[1];
  const float* cw    = (const float*)d_in[2];
  const float* cb    = (const float*)d_in[3];
  const float* Wx    = (const float*)d_in[4];
  const float* A_log = (const float*)d_in[5];
  const float* Dsk   = (const float*)d_in[6];
  const float* W_out = (const float*)d_in[7];
  const float* gamma = (const float*)d_in[8];
  const float* beta  = (const float*)d_in[9];
  float* out = (float*)d_out;

  char* p = (char*)d_ws;
  auto alloc = [&](size_t bytes) -> char* {
    char* r = p; p += (bytes + 255) & ~(size_t)255; return r;
  };
  _Float16* wint  = (_Float16*)alloc((size_t)4096 * 1024 * 2);   // W_in^T  f16
  _Float16* woutt = (_Float16*)alloc((size_t)1024 * 2048 * 2);   // W_out^T f16
  _Float16* xn16  = (_Float16*)alloc((size_t)ROWS * D_MODEL * 2);
  float* xz    = (float*)alloc((size_t)ROWS * 2 * D_INNER * 4);
  float* xconv = (float*)alloc((size_t)ROWS * D_INNER * 4);
  float* Bm    = (float*)alloc((size_t)ROWS * 16 * 4);
  float* Cm    = (float*)alloc((size_t)ROWS * 16 * 4);
  float* dl    = (float*)alloc((size_t)ROWS * 4);
  float* S     = (float*)alloc((size_t)BATCH * NCHUNK * D_INNER * 16 * 4);
  float* P     = (float*)alloc((size_t)BATCH * NCHUNK * D_INNER * 16 * 4);
  float* Hi    = (float*)alloc((size_t)BATCH * NCHUNK * D_INNER * 16 * 4);
  _Float16* yg16 = (_Float16*)alloc((size_t)ROWS * D_INNER * 2);

  convT<<<(1024 * 4096) / 256, 256, 0, stream>>>(W_in, wint, 1024, 4096);
  convT<<<(2048 * 1024) / 256, 256, 0, stream>>>(W_out, woutt, 2048, 1024);
  ln_to_f16<<<ROWS, 256, 0, stream>>>(x, gamma, beta, xn16);
  wmma_gemm<<<dim3(ROWS / 128, 4096 / 128), 256, 0, stream>>>(
      xn16, wint, xz, nullptr, ROWS, 4096, 1024);
  conv_silu<<<(ROWS * D_INNER) / 256, 256, 0, stream>>>(xz, cw, cb, xconv);
  proj_x<<<ROWS, 256, 0, stream>>>(xconv, Wx, Bm, Cm, dl);
  scan_chunk_summary<<<BATCH * NCHUNK * 8, 256, 0, stream>>>(xconv, Bm, dl, A_log, S, P);
  scan_chunk_prefix<<<(BATCH * D_INNER) / 256, 256, 0, stream>>>(S, P, Hi);
  scan_final<<<BATCH * NCHUNK * 8, 256, 0, stream>>>(
      xconv, xz, Bm, Cm, dl, A_log, Dsk, Hi, yg16);
  wmma_gemm<<<dim3(ROWS / 128, 1024 / 128), 256, 0, stream>>>(
      yg16, woutt, out, x, ROWS, 1024, 2048);
}
